// GCN_10264971838082
// MI455X (gfx1250) — matrix operand order repaired and checked by reading, compiled, not run on hardware
//
#include <hip/hip_runtime.h>

// GCN forward on MI455X (gfx1250, wave32).
//   layer1: K=2 matmul (VALU) -> aggregate (atomics) -> bias+relu+cvt bf16
//   layer2/3: bf16 WMMA GEMM [N,1024]x[1024,1024] -> aggregate -> bias(+relu)
//   mean-pool per graph -> tiny linear [1024,2]
// Roofline: message passing (~12 GB atomics traffic) bounds the pipeline; GEMMs
// run on v_wmma_f32_16x16x32_bf16 with B tiles staged into LDS by the Tensor
// Data Mover (tensor_load_to_lds, TENSORcnt) so the 8 waves of a block share
// one copy instead of 8 redundant L2 reads. K-loop fully unrolled so the
// accumulators stay in-place (no phi copies -> no WMMA hazard v_nops).

#define NN   100000
#define EE   400000
#define GG   128
#define HH   1024
#define CC   2
#define NPAD 100096   // multiple of 128 (GEMM M-tile), >= NN
#define KT   (HH / 32)

typedef __attribute__((ext_vector_type(16))) __bf16 bf16x16;
typedef __attribute__((ext_vector_type(8)))  __bf16 bf16x8;
typedef __attribute__((ext_vector_type(4)))  __bf16 bf16x4;
typedef __attribute__((ext_vector_type(8)))  float  floatx8;
typedef __attribute__((ext_vector_type(4)))  unsigned int u32x4;
typedef __attribute__((ext_vector_type(8)))  int i32x8;
typedef __attribute__((ext_vector_type(4)))  int i32x4;

// ---------------------------------------------------------------- utilities
__global__ void k_fill(float* __restrict__ p, float v, long n) {
    long i = (long)blockIdx.x * blockDim.x + threadIdx.x;
    if (i < n) p[i] = v;
}

// deg[c] += 1 for each edge target (deg pre-filled with 1.0 = self loop)
__global__ void k_deg(const int* __restrict__ col, float* __restrict__ deg) {
    int e = blockIdx.x * blockDim.x + threadIdx.x;
    if (e < EE) atomicAdd(&deg[col[e]], 1.0f);
}

__global__ void k_rsqrt(float* __restrict__ d) {
    int i = blockIdx.x * blockDim.x + threadIdx.x;
    if (i < NN) d[i] = rsqrtf(d[i]);   // deg >= 1 always (self loop)
}

// ---------------------------------------------------------------- layer 1 (K=2)
__global__ void k_lin1(const float* __restrict__ x, const float* __restrict__ W1,
                       float* __restrict__ h) {
    long i = (long)blockIdx.x * blockDim.x + threadIdx.x;     // over NN*HH/4
    if (i >= (long)NN * (HH / 4)) return;
    int n = (int)(i / (HH / 4));
    int j = (int)(i % (HH / 4)) * 4;
    float x0 = x[(size_t)n * 2 + 0];
    float x1 = x[(size_t)n * 2 + 1];
    float4 w0 = *(const float4*)&W1[j];        // W1 row 0
    float4 w1 = *(const float4*)&W1[HH + j];   // W1 row 1
    float4 o;
    o.x = x0 * w0.x + x1 * w1.x;
    o.y = x0 * w0.y + x1 * w1.y;
    o.z = x0 * w0.z + x1 * w1.z;
    o.w = x0 * w0.w + x1 * w1.w;
    *(float4*)&h[(size_t)n * HH + j] = o;
}

// ---------------------------------------------------------------- aggregation
// one block (256 threads) per edge: full 1024-wide row, float4 per thread
__global__ void k_agg(const float* __restrict__ h, const float* __restrict__ dis,
                      const int* __restrict__ row, const int* __restrict__ col,
                      float* __restrict__ agg) {
    __shared__ int   sr, sc;
    __shared__ float snrm;
    int e = blockIdx.x;                       // [0, EE+NN)
    if (threadIdx.x == 0) {
        int r, c; float nrm;
        if (e < EE) { r = row[e]; c = col[e]; nrm = dis[r] * dis[c]; }
        else        { r = c = e - EE; float d = dis[r]; nrm = d * d; }
        sr = r; sc = c; snrm = nrm;
    }
    __syncthreads();
    int r = sr, c = sc; float nrm = snrm;
    float4 v = ((const float4*)(h + (size_t)r * HH))[threadIdx.x];
    float* dst = agg + (size_t)c * HH + threadIdx.x * 4;
    atomicAdd(dst + 0, v.x * nrm);
    atomicAdd(dst + 1, v.y * nrm);
    atomicAdd(dst + 2, v.z * nrm);
    atomicAdd(dst + 3, v.w * nrm);
}

// ---------------------------------------------------------------- bias(+relu)+bf16
__global__ void k_bias_relu_bf16(const float* __restrict__ a, const float* __restrict__ b,
                                 __bf16* __restrict__ out, int do_relu) {
    long i = (long)blockIdx.x * blockDim.x + threadIdx.x;     // over NPAD*HH/4
    if (i >= (long)NPAD * (HH / 4)) return;
    int n = (int)(i / (HH / 4));
    int j = (int)(i % (HH / 4)) * 4;
    float4 v = make_float4(0.f, 0.f, 0.f, 0.f);
    if (n < NN) {
        v = *(const float4*)&a[(size_t)n * HH + j];
        float4 bb = *(const float4*)&b[j];
        v.x += bb.x; v.y += bb.y; v.z += bb.z; v.w += bb.w;
        if (do_relu) {
            v.x = fmaxf(v.x, 0.f); v.y = fmaxf(v.y, 0.f);
            v.z = fmaxf(v.z, 0.f); v.w = fmaxf(v.w, 0.f);
        }
    }
    bf16x4 o = { (__bf16)v.x, (__bf16)v.y, (__bf16)v.z, (__bf16)v.w };
    *(bf16x4*)&out[(size_t)n * HH + j] = o;
}

// layer-3 epilogue: fp32 out = agg + b3 (no relu, no cvt)
__global__ void k_bias_f32(const float* __restrict__ a, const float* __restrict__ b,
                           float* __restrict__ out) {
    long i = (long)blockIdx.x * blockDim.x + threadIdx.x;     // over NN*HH/4
    if (i >= (long)NN * (HH / 4)) return;
    int n = (int)(i / (HH / 4));
    int j = (int)(i % (HH / 4)) * 4;
    float4 v  = *(const float4*)&a[(size_t)n * HH + j];
    float4 bb = *(const float4*)&b[j];
    v.x += bb.x; v.y += bb.y; v.z += bb.z; v.w += bb.w;
    *(float4*)&out[(size_t)n * HH + j] = v;
}

// ---------------------------------------------------------------- weight repack
// Pack fp32 W[k][n] into per-lane B-fragment order (contiguous K per lane half, same as k_gemm's A):
// lanes 0-15 hold column (lane), K = kt*32 + 0..15; lanes 16-31 hold K = 16..31.
// packed index = ((nt*(HH/32) + kt)*32 + lane)*16 + e
__global__ void k_packW(const float* __restrict__ W, __bf16* __restrict__ wp) {
    int i = blockIdx.x * blockDim.x + threadIdx.x;            // over HH*HH
    if (i >= HH * HH) return;
    int e    = i & 15;
    int lane = (i >> 4) & 31;
    int kt   = (i >> 9) & (HH / 32 - 1);
    int nt   = i >> 14;
    int c = nt * 16 + (lane & 15);
    int k = kt * 32 + (lane >> 4) * 16 + e;
    wp[i] = (__bf16)W[(size_t)k * HH + c];
}

// ---------------------------------------------------------------- TDM helper
// Issue one tensor_load_to_lds of the 4-subtile B slice for K-step kt:
// 2-D tile, 4 rows x 1024 bytes, global row stride 32768 B (one packed subtile),
// data_size = 4 bytes -> tile 256 x 4 dwords, stride 8192 dwords.
// D# built per CDNA5 ISA.
__device__ __forceinline__ void tdm_load_b(const __bf16* Bp, int nt0, int kt,
                                           unsigned ldsOff) {
    unsigned long long ga = (unsigned long long)(size_t)Bp
                          + (unsigned long long)nt0 * 32768ull
                          + (unsigned long long)kt * 1024ull;
    u32x4 g0 = { 1u,                                   // count=1, user mode
                 ldsOff,                               // lds_addr (bytes)
                 (unsigned)ga,                         // global_addr[31:0]
                 (unsigned)((ga >> 32) & 0x1ffffffu)   // global_addr[56:32]
                   | 0x80000000u };                    // type=2 ("image")
    i32x8 g1 = { (int)(2u << 16),     // data_size=2 (4B), mask=0, no flags
                 (int)(256u << 16),   // tensor_dim0[15:0]=256 (bits 63:48)
                 (int)(4u << 16),     // tensor_dim0 hi=0, tensor_dim1=4
                 (int)(256u << 16),   // tensor_dim1 hi=0, tile_dim0=256
                 4,                   // tile_dim1=4, tile_dim2=0
                 8192,                // tensor_dim0_stride[31:0] (dwords)
                 0,                   // stride hi / dim1_stride lo
                 0 };
    i32x4 gz = { 0, 0, 0, 0 };        // groups 2/3: dims 2..4 unused
    asm volatile("tensor_load_to_lds %0, %1, %2, %3"
                 :: "s"(g0), "s"(g1), "s"(gz), "s"(gz)
                 : "memory");
}

// ---------------------------------------------------------------- WMMA GEMM
// C[NPAD,HH] = A[NPAD,HH](bf16) * Bp(packed bf16). Block = 8 waves = 128 rows
// x 64 cols; B staged per-block in LDS by the TDM (double-buffered), A streamed
// per-wave from global. K-loop (fully unrolled) of HH/32 steps, 4 WMMA/step.
__global__ void __launch_bounds__(256) k_gemm(const __bf16* __restrict__ A,
                                              const __bf16* __restrict__ Bp,
                                              float* __restrict__ Cout) {
    __shared__ __align__(32) __bf16 sB[2][4 * 512];    // 2 stages x 4KB

    const int lane  = threadIdx.x & 31;
    const int wave  = threadIdx.x >> 5;
    const int m0    = blockIdx.x * 128 + wave * 16;
    const int nt0   = blockIdx.y * 4;          // base 16-wide N subtile
    const int rowA  = m0 + (lane & 15);
    const int khalf = lane >> 4;               // 0: K 0..15, 1: K 16..31

    floatx8 acc0 = {}, acc1 = {}, acc2 = {}, acc3 = {};

    const __bf16* aBase = A + (size_t)rowA * HH + khalf * 16;
    const unsigned lds0 = (unsigned)(unsigned long long)(size_t)&sB[0][0];
    const unsigned lds1 = (unsigned)(unsigned long long)(size_t)&sB[1][0];

    // prologue: stage kt=0 into buffer 0
    if (threadIdx.x == 0) {
        tdm_load_b(Bp, nt0, 0, lds0);
        __builtin_amdgcn_s_wait_tensorcnt(0);
    }
    __syncthreads();

    #pragma unroll
    for (int kt = 0; kt < KT; ++kt) {
        const int cur = kt & 1;
        // kick off next stage (buffer 1-cur was released by last barrier)
        if (kt + 1 < KT) {
            if (threadIdx.x == 0)
                tdm_load_b(Bp, nt0, kt + 1, cur ? lds0 : lds1);
        }

        // A fragment: 16x32 bf16; per lane two contiguous 16B loads
        bf16x8 alo = *(const bf16x8*)(aBase + kt * 32);
        bf16x8 ahi = *(const bf16x8*)(aBase + kt * 32 + 8);
        bf16x16 a = __builtin_shufflevector(alo, ahi,
                        0,1,2,3,4,5,6,7,8,9,10,11,12,13,14,15);
        if (kt + 1 < KT)
            __builtin_prefetch(aBase + (kt + 1) * 32, 0, 0);   // global_prefetch_b8

        // B fragments from LDS (ds_load_b128 pairs)
        const __bf16* bl = &sB[cur][lane * 16];
        bf16x16 f0 = *(const bf16x16*)(bl + 0 * 512);
        bf16x16 f1 = *(const bf16x16*)(bl + 1 * 512);
        bf16x16 f2 = *(const bf16x16*)(bl + 2 * 512);
        bf16x16 f3 = *(const bf16x16*)(bl + 3 * 512);

        acc0 = __builtin_amdgcn_wmma_f32_16x16x32_bf16(false, a, false, f0, (short)0, acc0, false, false);
        acc1 = __builtin_amdgcn_wmma_f32_16x16x32_bf16(false, a, false, f1, (short)0, acc1, false, false);
        acc2 = __builtin_amdgcn_wmma_f32_16x16x32_bf16(false, a, false, f2, (short)0, acc2, false, false);
        acc3 = __builtin_amdgcn_wmma_f32_16x16x32_bf16(false, a, false, f3, (short)0, acc3, false, false);

        if (kt + 1 < KT) {
            if (threadIdx.x == 0)
                __builtin_amdgcn_s_wait_tensorcnt(0);   // next stage landed
            __syncthreads();                            // all done reading cur
        }
    }

    // C/D layout: VGPR r, lanes 0-15: M=r; lanes 16-31: M=r+8; N=lane%16
    const int mrow  = m0 + ((lane >> 4) << 3);
    const int ncol0 = nt0 * 16 + (lane & 15);
    #pragma unroll
    for (int r = 0; r < 8; ++r) {
        float* o = Cout + (size_t)(mrow + r) * HH + ncol0;
        o[0]  = acc0[r];
        o[16] = acc1[r];
        o[32] = acc2[r];
        o[48] = acc3[r];
    }
}

// ---------------------------------------------------------------- pooling + head
__global__ void k_cnt(const int* __restrict__ batch, float* __restrict__ cnt) {
    int n = blockIdx.x * blockDim.x + threadIdx.x;
    if (n < NN) atomicAdd(&cnt[batch[n]], 1.0f);
}

__global__ void k_pool(const float* __restrict__ h, const int* __restrict__ batch,
                       float* __restrict__ pooled) {
    int n = blockIdx.x;
    int g = batch[n];
    float4 v = ((const float4*)(h + (size_t)n * HH))[threadIdx.x];
    float* dst = pooled + (size_t)g * HH + threadIdx.x * 4;
    atomicAdd(dst + 0, v.x);
    atomicAdd(dst + 1, v.y);
    atomicAdd(dst + 2, v.z);
    atomicAdd(dst + 3, v.w);
}

__global__ void k_final(const float* __restrict__ pooled, const float* __restrict__ cnt,
                        const float* __restrict__ Wlin, const float* __restrict__ blin,
                        float* __restrict__ out) {
    int t = threadIdx.x;
    if (t >= GG * CC) return;
    int g = t / CC, c = t % CC;
    float s = 0.f;
    for (int j = 0; j < HH; ++j)
        s += pooled[(size_t)g * HH + j] * Wlin[(size_t)j * CC + c];
    float cn = cnt[g];
    cn = cn > 1.f ? cn : 1.f;
    out[t] = s / cn + blin[c];
}

// ---------------------------------------------------------------- driver
extern "C" void kernel_launch(void* const* d_in, const int* in_sizes, int n_in,
                              void* d_out, int out_size, void* d_ws, size_t ws_size,
                              hipStream_t stream) {
    (void)in_sizes; (void)n_in; (void)out_size; (void)ws_size;

    const float* x    = (const float*)d_in[0];
    const float* W1   = (const float*)d_in[1];
    const float* b1   = (const float*)d_in[2];
    const float* W2   = (const float*)d_in[3];
    const float* b2   = (const float*)d_in[4];
    const float* W3   = (const float*)d_in[5];
    const float* b3   = (const float*)d_in[6];
    const float* Wlin = (const float*)d_in[7];
    const float* blin = (const float*)d_in[8];
    const int*   ei   = (const int*)d_in[9];     // [2,E] flat
    const int*   bat  = (const int*)d_in[10];
    float*       out  = (float*)d_out;

    const int* row = ei;
    const int* col = ei + EE;

    // workspace carve-up (256B aligned)
    char* ws = (char*)d_ws;
    size_t off = 0;
    auto carve = [&](size_t bytes) {
        void* p = ws + off;
        off = (off + bytes + 255) & ~(size_t)255;
        return p;
    };
    float*  buf_h  = (float*) carve((size_t)NPAD * HH * 4);
    float*  buf_a  = (float*) carve((size_t)NPAD * HH * 4);
    __bf16* buf_bf = (__bf16*)carve((size_t)NPAD * HH * 2);
    float*  dis    = (float*) carve((size_t)NN * 4);
    __bf16* wp2    = (__bf16*)carve((size_t)HH * HH * 2);
    __bf16* wp3    = (__bf16*)carve((size_t)HH * HH * 2);
    float*  pooled = (float*) carve((size_t)GG * HH * 4);
    float*  cnt    = (float*) carve((size_t)GG * 4);

    const long nh   = (long)NN * HH;
    dim3 blk(256);
    auto grid1 = [](long n) { return dim3((unsigned)((n + 255) / 256)); };

    // degrees & symmetric norm
    k_fill <<<grid1(NN), blk, 0, stream>>>(dis, 1.0f, NN);           // self loop
    k_deg  <<<grid1(EE), blk, 0, stream>>>(col, dis);
    k_rsqrt<<<grid1(NN), blk, 0, stream>>>(dis);

    // pack weights for the WMMA B-fragment layout
    k_packW<<<grid1((long)HH * HH), blk, 0, stream>>>(W2, wp2);
    k_packW<<<grid1((long)HH * HH), blk, 0, stream>>>(W3, wp3);

    // ---- layer 1: h = x@W1 ; agg ; relu(agg+b1) -> bf16
    k_lin1<<<grid1(nh / 4), blk, 0, stream>>>(x, W1, buf_h);
    k_fill<<<grid1(nh), blk, 0, stream>>>(buf_a, 0.0f, nh);
    k_agg <<<dim3(EE + NN), blk, 0, stream>>>(buf_h, dis, row, col, buf_a);
    k_bias_relu_bf16<<<grid1((long)NPAD * HH / 4), blk, 0, stream>>>(buf_a, b1, buf_bf, 1);

    // ---- layer 2: WMMA GEMM ; agg ; relu(agg+b2) -> bf16
    k_gemm<<<dim3(NPAD / 128, HH / 64), blk, 0, stream>>>(buf_bf, wp2, buf_h);
    k_fill<<<grid1(nh), blk, 0, stream>>>(buf_a, 0.0f, nh);
    k_agg <<<dim3(EE + NN), blk, 0, stream>>>(buf_h, dis, row, col, buf_a);
    k_bias_relu_bf16<<<grid1((long)NPAD * HH / 4), blk, 0, stream>>>(buf_a, b2, buf_bf, 1);

    // ---- layer 3: WMMA GEMM ; agg ; agg+b3 (fp32, no relu)
    k_gemm<<<dim3(NPAD / 128, HH / 64), blk, 0, stream>>>(buf_bf, wp3, buf_h);
    k_fill<<<grid1(nh), blk, 0, stream>>>(buf_a, 0.0f, nh);
    k_agg <<<dim3(EE + NN), blk, 0, stream>>>(buf_h, dis, row, col, buf_a);
    k_bias_f32<<<grid1(nh / 4), blk, 0, stream>>>(buf_a, b3, buf_h);

    // ---- mean pool + linear head
    k_fill <<<grid1((long)GG * HH), blk, 0, stream>>>(pooled, 0.0f, (long)GG * HH);
    k_fill <<<grid1(GG), blk, 0, stream>>>(cnt, 0.0f, GG);
    k_cnt  <<<grid1(NN), blk, 0, stream>>>(bat, cnt);
    k_pool <<<dim3(NN), blk, 0, stream>>>(buf_h, bat, pooled);
    k_final<<<dim3(1), blk, 0, stream>>>(pooled, cnt, Wlin, blin, out);
}
